// MatchingNetwork_86577950753065
// MI455X (gfx1250) — matrix operand order repaired and checked by reading, not compiled, not run on hardware
//
#include <hip/hip_runtime.h>

// ---------------------------------------------------------------------------
// MatchingNetwork forward on gfx1250 (MI455X, wave32, WMMA).
// conv2/3/4 are implicit GEMMs (M=64, K=576, N=B*OH*OW) on
// v_wmma_f32_16x16x32_bf16. One wave computes the full M=64 x 16 column
// block for its N-tile: B-fragments (the dominant memory stream) are loaded
// once per K-step and reused across 4 independent WMMA accumulation chains.
// ---------------------------------------------------------------------------

typedef __attribute__((ext_vector_type(16))) __bf16        v16bf;
typedef __attribute__((ext_vector_type(8)))  float         v8f;
typedef __attribute__((ext_vector_type(8)))  unsigned int  v8u;

#define BN_EPS 1e-5f
#define KSTEPS 18   // K = 576 = 18 * 32

__device__ __forceinline__ unsigned short f32_bf16(float f) {
  unsigned int u = __float_as_uint(f);
  u += 0x7FFFu + ((u >> 16) & 1u);          // round-to-nearest-even
  return (unsigned short)(u >> 16);
}

// ---------------- conv1: in-ch 1, K=9 -> plain VALU ------------------------
__global__ void conv1_kernel(const float* __restrict__ x, long xoff, long istride,
                             const float* __restrict__ w, const float* __restrict__ b,
                             float* __restrict__ out) {
  int idx = blockIdx.x * blockDim.x + threadIdx.x;      // 64*64*676
  if (idx >= 64 * 64 * 676) return;
  int sp  = idx % 676;
  int oc  = (idx / 676) & 63;
  int img = idx / (676 * 64);
  int oh = sp / 26, ow = sp % 26;
  const float* xi = x + xoff + (long)img * istride;
  const float* wk = w + oc * 9;
  float acc = b[oc];
#pragma unroll
  for (int kh = 0; kh < 3; ++kh)
#pragma unroll
    for (int kw = 0; kw < 3; ++kw)
      acc += xi[(oh + kh) * 28 + ow + kw] * wk[kh * 3 + kw];
  out[idx] = acc;   // layout [img][oc][26][26]
}

// ---------------- per-channel batch stats (mean, rsqrt(var+eps)) -----------
__global__ void stats_kernel(const float* __restrict__ act, int HW,
                             float* __restrict__ stats) {
  __shared__ float s1[256], s2[256];
  int c = blockIdx.x, tid = threadIdx.x;
  float sum = 0.f, sq = 0.f;
  for (int b = 0; b < 64; ++b) {
    const float* p = act + (size_t)(b * 64 + c) * HW;
    for (int i = tid; i < HW; i += 256) { float v = p[i]; sum += v; sq += v * v; }
  }
  s1[tid] = sum; s2[tid] = sq; __syncthreads();
  for (int s = 128; s > 0; s >>= 1) {
    if (tid < s) { s1[tid] += s1[tid + s]; s2[tid] += s2[tid + s]; }
    __syncthreads();
  }
  if (tid == 0) {
    float cnt = 64.f * (float)HW;
    float m   = s1[0] / cnt;
    float var = s2[0] / cnt - m * m;
    stats[2 * c]     = m;
    stats[2 * c + 1] = rsqrtf(var + BN_EPS);
  }
}

// ---------------- weight -> A-fragment (16x32 bf16 WMMA layout) ------------
// Afrag dword index: ((mtile*18 + ks)*32 + lane)*8 + j
__global__ void afrag_kernel(const float* __restrict__ w, unsigned int* __restrict__ Afrag) {
  int idx = blockIdx.x * blockDim.x + threadIdx.x;      // 4*18*256 = 18432
  if (idx >= 18432) return;
  int j     = idx & 7;
  int lane  = (idx >> 3) & 31;
  int ks    = (idx >> 8) % KSTEPS;
  int mtile = idx / (KSTEPS * 256);
  int m = mtile * 16 + (lane & 15);
  unsigned int pack = 0;
#pragma unroll
  for (int t = 0; t < 2; ++t) {
    int klocal = ((j < 4) ? 2 * j : 16 + 2 * (j - 4)) + ((lane < 16) ? 0 : 8) + t;
    int k  = ks * 32 + klocal;                           // 0..575
    int ic = k / 9, r = k % 9;
    float v = w[(m * 64 + ic) * 9 + r];                  // OIHW
    pack |= ((unsigned int)f32_bf16(v)) << (16 * t);
  }
  Afrag[idx] = pack;
}

// ---------------- BN+ReLU + im2col into B-fragment (32x16 bf16 layout) -----
// Bfrag dword index: ((ntile*18 + ks)*32 + lane)*8 + j
__global__ void im2col_frag_kernel(const float* __restrict__ act,
                                   const float* __restrict__ stats,
                                   const float* __restrict__ gamma,
                                   const float* __restrict__ beta,
                                   unsigned int* __restrict__ Bfrag,
                                   int IH, int IW, int OH, int OW, int OHW,
                                   int total, int applyBN) {
  int idx = blockIdx.x * blockDim.x + threadIdx.x;
  if (idx >= total) return;
  int j     = idx & 7;
  int lane  = (idx >> 3) & 31;
  int ks    = (idx >> 8) % KSTEPS;
  int ntile = idx / (KSTEPS * 256);
  int n   = ntile * 16 + (lane & 15);
  int img = n / OHW, sp = n % OHW;
  int oh = sp / OW, ow = sp % OW;
  unsigned int pack = 0;
#pragma unroll
  for (int t = 0; t < 2; ++t) {
    int klocal = ((j < 4) ? 2 * j : 16 + 2 * (j - 4)) + ((lane < 16) ? 0 : 8) + t;
    int k  = ks * 32 + klocal;                           // 0..575
    int ic = k / 9, r = k % 9;
    int kh = r / 3, kw = r % 3;
    float v = act[((size_t)(img * 64 + ic) * IH + (oh + kh)) * IW + (ow + kw)];
    if (applyBN) {
      v = (v - stats[2 * ic]) * stats[2 * ic + 1] * gamma[ic] + beta[ic];
      v = fmaxf(v, 0.f);
    }
    pack |= ((unsigned int)f32_bf16(v)) << (16 * t);
  }
  Bfrag[idx] = pack;
}

// ---------------- WMMA GEMM: one wave = full M=64 x N=16 column block ------
// 4 independent 16x16 accumulators; B loaded once per K-step, reused 4x.
__global__ void wmma_gemm_kernel(const unsigned int* __restrict__ Afrag,
                                 const unsigned int* __restrict__ Bfrag,
                                 const float* __restrict__ bias,
                                 float* __restrict__ out, int OHW) {
  int ntile = blockIdx.x;
  int lane  = threadIdx.x;                               // blockDim.x == 32
  const v8u* Ap = (const v8u*)Afrag + lane;              // + (mtile*18+ks)*32
  const v8u* Bp = (const v8u*)Bfrag + (size_t)(ntile * KSTEPS) * 32 + lane;
  v8f acc[4];
#pragma unroll
  for (int mt = 0; mt < 4; ++mt) acc[mt] = (v8f){0.f,0.f,0.f,0.f,0.f,0.f,0.f,0.f};
#pragma unroll
  for (int ks = 0; ks < KSTEPS; ++ks) {
    union { v8u u; v16bf b; } bfr;
    bfr.u = Bp[ks * 32];
#pragma unroll
    for (int mt = 0; mt < 4; ++mt) {
      union { v8u u; v16bf b; } afr;
      afr.u = Ap[(mt * KSTEPS + ks) * 32];
      acc[mt] = __builtin_amdgcn_wmma_f32_16x16x32_bf16(
          /*neg_a=*/false, afr.b, /*neg_b=*/false, bfr.b,
          /*c_mod=*/(short)0, acc[mt], /*reuse_a=*/false, /*reuse_b=*/false);
    }
  }
  // D layout: VGPR r -> lanes 0-15: (M=r, N=lane); lanes 16-31: (M=r+8, N=lane-16)
  int nl   = lane & 15;
  int moff = (lane < 16) ? 0 : 8;
  int n    = ntile * 16 + nl;
  int img  = n / OHW, sp = n % OHW;
  float* obase = out + (size_t)img * 64 * OHW + sp;
#pragma unroll
  for (int mt = 0; mt < 4; ++mt) {
#pragma unroll
    for (int r = 0; r < 8; ++r) {
      int oc = mt * 16 + moff + r;
      obase[(size_t)oc * OHW] = acc[mt][r] + bias[oc];
    }
  }
}

// ---------------- BN + ReLU + 2x2 maxpool ----------------------------------
__global__ void bn_relu_pool_kernel(const float* __restrict__ act,
                                    const float* __restrict__ stats,
                                    const float* __restrict__ gamma,
                                    const float* __restrict__ beta,
                                    float* __restrict__ out, int H, int W) {
  int PH = H >> 1, PW = W >> 1;
  int total = 64 * 64 * PH * PW;
  int idx = blockIdx.x * blockDim.x + threadIdx.x;
  if (idx >= total) return;
  int pw  = idx % PW;
  int ph  = (idx / PW) % PH;
  int c   = (idx / (PW * PH)) & 63;
  int img = idx / (PW * PH * 64);
  const float* p = act + ((size_t)(img * 64 + c) * H + 2 * ph) * W + 2 * pw;
  float m = stats[2 * c], rs = stats[2 * c + 1], g = gamma[c], bt = beta[c];
  float best = -1e30f;
#pragma unroll
  for (int dh = 0; dh < 2; ++dh)
#pragma unroll
    for (int dw = 0; dw < 2; ++dw) {
      float v = (p[dh * W + dw] - m) * rs * g + bt;
      v = fmaxf(v, 0.f);
      best = fmaxf(best, v);
    }
  out[idx] = best;    // [img][c][PH][PW]; for 8x8->4x4 this is exactly flatten order
}

// ---------------- similarity + softmax + argmax + loss ---------------------
__global__ void final_kernel(const float* __restrict__ emb,
                             const int* __restrict__ ty,
                             float* __restrict__ out) {
  __shared__ float red[128];
  __shared__ float z[20];
  int b = blockIdx.x, tid = threadIdx.x;                 // 64 blocks x 128 thr
  const float* et = emb + (size_t)(20 * 64 + b) * 1024;  // target embedding
  for (int n = 0; n < 20; ++n) {
    const float* es = emb + (size_t)(n * 64 + b) * 1024;
    float d = 0.f, q = 0.f;
    for (int i = tid; i < 1024; i += 128) {
      float s = es[i];
      d += s * et[i];
      q += s * s;
    }
    red[tid] = d; __syncthreads();
    for (int s = 64; s > 0; s >>= 1) { if (tid < s) red[tid] += red[tid + s]; __syncthreads(); }
    float dot = red[0]; __syncthreads();
    red[tid] = q; __syncthreads();
    for (int s = 64; s > 0; s >>= 1) { if (tid < s) red[tid] += red[tid + s]; __syncthreads(); }
    if (tid == 0) z[n] = dot * rsqrtf(red[0]);
    __syncthreads();
  }
  if (tid == 0) {
    float mx = -1e30f; int am = 0;
    for (int n = 0; n < 20; ++n) if (z[n] > mx) { mx = z[n]; am = n; }
    float se = 0.f;
    for (int n = 0; n < 20; ++n) se += expf(z[n] - mx);
    float p[20];
    for (int n = 0; n < 20; ++n) p[n] = expf(z[n] - mx) / se;
    // log_softmax applied to the probabilities (faithful double softmax)
    float mp = -1e30f;
    for (int n = 0; n < 20; ++n) mp = fmaxf(mp, p[n]);
    float se2 = 0.f;
    for (int n = 0; n < 20; ++n) se2 += expf(p[n] - mp);
    int y = ty[b];
    float lp = p[y] - mp - logf(se2);
    atomicAdd(&out[0], ((am == y) ? 1.f : 0.f) / 64.f);
    atomicAdd(&out[1], (-lp) / 64.f);
  }
}

__global__ void zero2_kernel(float* out) {
  if (threadIdx.x < 2) out[threadIdx.x] = 0.f;
}

// ---------------------------------------------------------------------------
extern "C" void kernel_launch(void* const* d_in, const int* in_sizes, int n_in,
                              void* d_out, int out_size, void* d_ws, size_t ws_size,
                              hipStream_t stream) {
  const float* support = (const float*)d_in[0];   // [64,20,1,28,28]
  const float* target  = (const float*)d_in[1];   // [64,1,28,28]
  const int*   ty      = (const int*)d_in[2];     // [64]
  const float* w1 = (const float*)d_in[3];  const float* b1 = (const float*)d_in[4];
  const float* w2 = (const float*)d_in[5];  const float* b2 = (const float*)d_in[6];
  const float* w3 = (const float*)d_in[7];  const float* b3 = (const float*)d_in[8];
  const float* w4 = (const float*)d_in[9];  const float* b4 = (const float*)d_in[10];
  const float* gamma = (const float*)d_in[11];
  const float* beta  = (const float*)d_in[12];
  float* out = (float*)d_out;

  // ---- workspace carve-up (all fully written before read each call) ----
  char* ws = (char*)d_ws;
  size_t off = 0;
  auto alloc = [&](size_t bytes) -> void* {
    void* p = ws + off;
    off += (bytes + 255) & ~(size_t)255;
    return p;
  };
  float*        emb   = (float*)alloc((size_t)21 * 64 * 1024 * 4);  // 21 calls x [64][1024]
  unsigned int* Af2   = (unsigned int*)alloc(18432 * 4);
  unsigned int* Af3   = (unsigned int*)alloc(18432 * 4);
  unsigned int* Af4   = (unsigned int*)alloc(18432 * 4);
  float*        stats = (float*)alloc(64 * 2 * 4);
  float*        bufA  = (float*)alloc((size_t)64 * 64 * 676 * 4);   // conv1 / conv3 out
  float*        bufB  = (float*)alloc((size_t)64 * 64 * 576 * 4);   // conv2 / conv4 out
  float*        bufP  = (float*)alloc((size_t)64 * 64 * 144 * 4);   // pool2 out
  unsigned int* Bfrag = (unsigned int*)alloc((size_t)2304 * KSTEPS * 256 * 4);
  (void)ws_size; (void)in_sizes; (void)n_in; (void)out_size;

  // ---- weight fragments (recomputed every launch: deterministic) ----
  afrag_kernel<<<(18432 + 255) / 256, 256, 0, stream>>>(w2, Af2);
  afrag_kernel<<<(18432 + 255) / 256, 256, 0, stream>>>(w3, Af3);
  afrag_kernel<<<(18432 + 255) / 256, 256, 0, stream>>>(w4, Af4);
  zero2_kernel<<<1, 32, 0, stream>>>(out);

  const int tot2 = 2304 * KSTEPS * 256;   // N=36864
  const int tot3 =  400 * KSTEPS * 256;   // N=6400
  const int tot4 =  256 * KSTEPS * 256;   // N=4096

  for (int call = 0; call < 21; ++call) {
    const float* x; long xoff, istride;
    if (call < 20) { x = support; xoff = (long)call * 784; istride = 20L * 784; }
    else           { x = target;  xoff = 0;                istride = 784; }

    // conv1 -> bufA [64,64,26,26]
    conv1_kernel<<<(64 * 64 * 676 + 255) / 256, 256, 0, stream>>>(
        x, xoff, istride, w1, b1, bufA);
    stats_kernel<<<64, 256, 0, stream>>>(bufA, 676, stats);

    // BN+ReLU+im2col -> conv2 WMMA -> bufB [64,64,24,24]
    im2col_frag_kernel<<<(tot2 + 255) / 256, 256, 0, stream>>>(
        bufA, stats, gamma, beta, Bfrag, 26, 26, 24, 24, 576, tot2, 1);
    wmma_gemm_kernel<<<dim3(2304), 32, 0, stream>>>(Af2, Bfrag, b2, bufB, 576);
    stats_kernel<<<64, 256, 0, stream>>>(bufB, 576, stats);

    // BN+ReLU+pool -> bufP [64,64,12,12]
    bn_relu_pool_kernel<<<(64 * 64 * 144 + 255) / 256, 256, 0, stream>>>(
        bufB, stats, gamma, beta, bufP, 24, 24);

    // im2col (already normalized) -> conv3 WMMA -> bufA [64,64,10,10]
    im2col_frag_kernel<<<(tot3 + 255) / 256, 256, 0, stream>>>(
        bufP, stats, gamma, beta, Bfrag, 12, 12, 10, 10, 100, tot3, 0);
    wmma_gemm_kernel<<<dim3(400), 32, 0, stream>>>(Af3, Bfrag, b3, bufA, 100);
    stats_kernel<<<64, 256, 0, stream>>>(bufA, 100, stats);

    // BN+ReLU+im2col -> conv4 WMMA -> bufB [64,64,8,8]
    im2col_frag_kernel<<<(tot4 + 255) / 256, 256, 0, stream>>>(
        bufA, stats, gamma, beta, Bfrag, 10, 10, 8, 8, 64, tot4, 1);
    wmma_gemm_kernel<<<dim3(256), 32, 0, stream>>>(Af4, Bfrag, b4, bufB, 64);
    stats_kernel<<<64, 256, 0, stream>>>(bufB, 64, stats);

    // BN+ReLU+pool -> embedding [64][1024] (flatten order c*16+h*4+w)
    bn_relu_pool_kernel<<<(64 * 64 * 16 + 255) / 256, 256, 0, stream>>>(
        bufB, stats, gamma, beta, emb + (size_t)call * 64 * 1024, 8, 8);
  }

  final_kernel<<<64, 128, 0, stream>>>(emb, ty, out);
}
